// DAD_84507776516366
// MI455X (gfx1250) — matrix operand order repaired and checked by reading, not compiled, not run on hardware
//
#include <hip/hip_runtime.h>

// ---------------------------------------------------------------------------
// Problem constants (from reference): B=2048, M=512, A=2048, S=3*A=6144
// ---------------------------------------------------------------------------
#define BATCH 2048
#define MDIM  512
#define ADIM  2048
#define SDIM  6144
#define RHO_C   1.0f
#define THRESH  0.1f   // LAMDA / RHO
#define ITERS 10

typedef __attribute__((ext_vector_type(16))) __bf16 v16bf;
typedef __attribute__((ext_vector_type(8)))  float  v8f;

union FragBF { v16bf v; uint4 q[2]; };

#if __has_builtin(__builtin_amdgcn_global_load_async_to_lds_b128)
#define HAVE_ASYNC_LDS 1
// Exact param types per hipcc diagnostic: int __vector(4) pointers in AS1/AS3
typedef int v4ig __attribute__((vector_size(16)));
typedef __attribute__((address_space(1))) v4ig* g_v4i_t;
typedef __attribute__((address_space(3))) v4ig* l_v4i_t;
#define ASYNC_G(p) ((g_v4i_t)(void*)(p))
#define ASYNC_L(p) ((l_v4i_t)(void*)(p))
#else
#define HAVE_ASYNC_LDS 0
#endif

static __device__ __forceinline__ void async_wait0() {
#if __has_builtin(__builtin_amdgcn_s_wait_asynccnt)
    __builtin_amdgcn_s_wait_asynccnt(0);
#else
    asm volatile("s_wait_asynccnt 0x0" ::: "memory");
#endif
}

static __device__ __forceinline__ __bf16 f2bf(float f) {
    unsigned u; __builtin_memcpy(&u, &f, 4);
    unsigned r = u + 0x7FFFu + ((u >> 16) & 1u);   // round-to-nearest-even
    unsigned short h = (unsigned short)(r >> 16);
    __bf16 b; __builtin_memcpy(&b, &h, 2);
    return b;
}

// ---------------------------------------------------------------------------
// Elementwise conversion / init kernels (grid-stride)
// ---------------------------------------------------------------------------
__global__ void k_cvt_bf16(const float* __restrict__ in, __bf16* __restrict__ out, int n) {
    for (int i = blockIdx.x * blockDim.x + threadIdx.x; i < n; i += gridDim.x * blockDim.x)
        out[i] = f2bf(in[i]);
}

__global__ void k_zero_f32(float* __restrict__ p, int n) {
    for (int i = blockIdx.x * blockDim.x + threadIdx.x; i < n; i += gridDim.x * blockDim.x)
        p[i] = 0.0f;
}

__global__ void k_zero_bf16(__bf16* __restrict__ p, int n) {
    __bf16 z = f2bf(0.0f);
    for (int i = blockIdx.x * blockDim.x + threadIdx.x; i < n; i += gridDim.x * blockDim.x)
        p[i] = z;
}

// Transposing f32 -> bf16 convert: in is R x C row-major, out is C x R row-major.
__global__ void k_transpose_bf16(const float* __restrict__ in, __bf16* __restrict__ out,
                                 int R, int C) {
    __shared__ float tile[32][33];
    int rb = blockIdx.y * 32, cb = blockIdx.x * 32;
    for (int i = threadIdx.y; i < 32; i += 8)
        tile[i][threadIdx.x] = in[(size_t)(rb + i) * C + (cb + threadIdx.x)];
    __syncthreads();
    for (int i = threadIdx.y; i < 32; i += 8)
        out[(size_t)(cb + i) * R + (rb + threadIdx.x)] = f2bf(tile[threadIdx.x][i]);
}

// ---------------------------------------------------------------------------
// min/max reduction of x (two stage)
// ---------------------------------------------------------------------------
__global__ void k_minmax_part(const float* __restrict__ x, int n,
                              float* __restrict__ pmin, float* __restrict__ pmax) {
    __shared__ float smn[256], smx[256];
    int tid = threadIdx.x;
    float mn = 3.402823e38f, mx = -3.402823e38f;
    for (int i = blockIdx.x * 256 + tid; i < n; i += gridDim.x * 256) {
        float v = x[i];
        mn = fminf(mn, v); mx = fmaxf(mx, v);
    }
    smn[tid] = mn; smx[tid] = mx;
    __syncthreads();
    for (int s = 128; s > 0; s >>= 1) {
        if (tid < s) { smn[tid] = fminf(smn[tid], smn[tid + s]); smx[tid] = fmaxf(smx[tid], smx[tid + s]); }
        __syncthreads();
    }
    if (tid == 0) { pmin[blockIdx.x] = smn[0]; pmax[blockIdx.x] = smx[0]; }
}

__global__ void k_minmax_final(const float* __restrict__ pmin, const float* __restrict__ pmax,
                               int nb, float* __restrict__ mnmx) {
    __shared__ float smn[256], smx[256];
    int tid = threadIdx.x;
    float mn = 3.402823e38f, mx = -3.402823e38f;
    for (int i = tid; i < nb; i += 256) { mn = fminf(mn, pmin[i]); mx = fmaxf(mx, pmax[i]); }
    smn[tid] = mn; smx[tid] = mx;
    __syncthreads();
    for (int s = 128; s > 0; s >>= 1) {
        if (tid < s) { smn[tid] = fminf(smn[tid], smn[tid + s]); smx[tid] = fmaxf(smx[tid], smx[tid + s]); }
        __syncthreads();
    }
    if (tid == 0) { mnmx[0] = smn[0]; mnmx[1] = smx[0]; }
}

// ---------------------------------------------------------------------------
// Unpivoted right-looking LU; only diag(U) is needed -> dU[k] = 1/U[k][k].
// Single workgroup (setup cost, one-time).
// ---------------------------------------------------------------------------
__global__ __launch_bounds__(1024)
void k_lu_diag(float* __restrict__ m, float* __restrict__ dU, int n) {
    int tid = threadIdx.x;
    for (int k = 0; k < n; ++k) {
        float piv = m[(size_t)k * n + k];
        if (tid == 0) dU[k] = 1.0f / piv;
        __syncthreads();
        for (int i = k + 1 + tid; i < n; i += 1024) {
            float l = m[(size_t)i * n + k] / piv;
            const float* rk = m + (size_t)k * n;
            float* ri = m + (size_t)i * n;
            for (int j = k + 1; j < n; ++j) ri[j] -= l * rk[j];
        }
        __syncthreads();
    }
}

// ---------------------------------------------------------------------------
// Generic WMMA GEMM:  acc = A(MxK bf16, row-major) * Bt(NxK bf16, row-major)^T
// Block tile 128x128x32, 256 threads = 8 wave32 waves in 2x4, each wave 64x32.
// Double-buffered LDS tiles filled by GLOBAL_LOAD_ASYNC_TO_LDS_B128 (ASYNCcnt)
// when the toolchain exposes the builtin; sync fallback otherwise.
// MODE 0: C = alpha*acc (+C if beta)            (f32 C, MxN)
// MODE 1: xh = dU[col]*(x0 + alpha*acc); outf=xh; outbf=bf16(xh)
// MODE 2: fxu = acc + u; z = shrink(fxu, 0.1); u += fxu - z; outbf = bf16(z-u_new)
// All dims must be multiples of 128 (M,N) and 32 (K) -- true for this problem.
// ---------------------------------------------------------------------------
template <int MODE>
__global__ __launch_bounds__(256)
void k_gemm(const __bf16* __restrict__ Abuf, const __bf16* __restrict__ Btbuf,
            float* __restrict__ C, const float* __restrict__ x0,
            const float* __restrict__ dU, float* __restrict__ outf,
            __bf16* __restrict__ outbf, float* __restrict__ u,
            int Mdim, int Ndim, int Kdim, float alpha, int beta) {
    // padded pitch (40 bf16 = 80B) -> 16-lane fragment reads hit distinct banks
    __shared__ __align__(16) __bf16 Atile[2][128][40];
    __shared__ __align__(16) __bf16 Btile[2][128][40];

    const int tid   = threadIdx.x;
    const int lane  = tid & 31;
    const int wave  = tid >> 5;
    const int wr    = wave >> 2;        // 0..1  (64 rows each)
    const int wc    = wave & 3;         // 0..3  (32 cols each)
    const int lhalf = lane >> 4;        // 0/1
    const int l15   = lane & 15;
    const int m0    = blockIdx.y * 128;
    const int n0    = blockIdx.x * 128;

    // each thread owns two 16B chunks per matrix: ids tid and tid+256
    const int r0  = tid >> 2,         cc0 = (tid & 3) << 3;
    const int r1  = (tid + 256) >> 2, cc1 = ((tid + 256) & 3) << 3;

    v8f acc[4][2];
    #pragma unroll
    for (int i = 0; i < 4; ++i)
        #pragma unroll
        for (int j = 0; j < 2; ++j)
            #pragma unroll
            for (int e = 0; e < 8; ++e) acc[i][j][e] = 0.0f;

    const int ah = lhalf << 3;          // A frag: K chunks {ah..ah+7, ah+16..ah+23}
    const int bk = lhalf << 3;          // B frag: K chunks {bk..bk+7, bk+16..bk+23}

    auto stage = [&](int buf, int k0) {
#if HAVE_ASYNC_LDS
        __builtin_amdgcn_global_load_async_to_lds_b128(
            ASYNC_G(Abuf + (size_t)(m0 + r0) * Kdim + (k0 + cc0)), ASYNC_L(&Atile[buf][r0][cc0]), 0, 0);
        __builtin_amdgcn_global_load_async_to_lds_b128(
            ASYNC_G(Abuf + (size_t)(m0 + r1) * Kdim + (k0 + cc1)), ASYNC_L(&Atile[buf][r1][cc1]), 0, 0);
        __builtin_amdgcn_global_load_async_to_lds_b128(
            ASYNC_G(Btbuf + (size_t)(n0 + r0) * Kdim + (k0 + cc0)), ASYNC_L(&Btile[buf][r0][cc0]), 0, 0);
        __builtin_amdgcn_global_load_async_to_lds_b128(
            ASYNC_G(Btbuf + (size_t)(n0 + r1) * Kdim + (k0 + cc1)), ASYNC_L(&Btile[buf][r1][cc1]), 0, 0);
#else
        uint4 a0 = *(const uint4*)(Abuf + (size_t)(m0 + r0) * Kdim + (k0 + cc0));
        uint4 a1 = *(const uint4*)(Abuf + (size_t)(m0 + r1) * Kdim + (k0 + cc1));
        uint4 b0 = *(const uint4*)(Btbuf + (size_t)(n0 + r0) * Kdim + (k0 + cc0));
        uint4 b1 = *(const uint4*)(Btbuf + (size_t)(n0 + r1) * Kdim + (k0 + cc1));
        *(uint4*)(&Atile[buf][r0][cc0]) = a0;
        *(uint4*)(&Atile[buf][r1][cc1]) = a1;
        *(uint4*)(&Btile[buf][r0][cc0]) = b0;
        *(uint4*)(&Btile[buf][r1][cc1]) = b1;
#endif
    };

    const int nk = Kdim >> 5;           // K-tiles of 32
    stage(0, 0);
#if HAVE_ASYNC_LDS
    async_wait0();
#endif
    __syncthreads();

    for (int kt = 0; kt < nk; ++kt) {
        const int buf = kt & 1;
        if (kt + 1 < nk) stage(buf ^ 1, (kt + 1) << 5);   // async-fill next buffer

        // ---- fragments (ISA 16-bit A 16x32 / B 32x16 wave32 layouts) + WMMA ----
        FragBF af[4], bfr[2];
        #pragma unroll
        for (int i = 0; i < 4; ++i) {
            const __bf16* p = &Atile[buf][wr * 64 + i * 16 + l15][0];
            af[i].q[0] = *(const uint4*)(p + ah);
            af[i].q[1] = *(const uint4*)(p + ah + 16);
        }
        #pragma unroll
        for (int j = 0; j < 2; ++j) {
            const __bf16* p = &Btile[buf][wc * 32 + j * 16 + l15][0];
            bfr[j].q[0] = *(const uint4*)(p + bk);
            bfr[j].q[1] = *(const uint4*)(p + bk + 16);
        }
        #pragma unroll
        for (int i = 0; i < 4; ++i)
            #pragma unroll
            for (int j = 0; j < 2; ++j)
                acc[i][j] = __builtin_amdgcn_wmma_f32_16x16x32_bf16(
                    false, af[i].v, false, bfr[j].v, (short)0, acc[i][j], false, false);

#if HAVE_ASYNC_LDS
        async_wait0();                  // my async writes for next buffer landed
#endif
        __syncthreads();                // everyone done reading buf / writing buf^1
    }

    // ---- fused epilogue; C/D layout: VGPR e -> row e + 8*lhalf, col = l15 ----
    #pragma unroll
    for (int i = 0; i < 4; ++i) {
        #pragma unroll
        for (int j = 0; j < 2; ++j) {
            #pragma unroll
            for (int e = 0; e < 8; ++e) {
                int row = m0 + wr * 64 + i * 16 + (lhalf << 3) + e;
                int col = n0 + wc * 32 + j * 16 + l15;
                float val = acc[i][j][e] * alpha;
                size_t idx = (size_t)row * Ndim + col;
                if (MODE == 0) {
                    C[idx] = beta ? (C[idx] + val) : val;
                } else if (MODE == 1) {
                    float xh = dU[col] * (x0[idx] + val);
                    outf[idx]  = xh;
                    outbf[idx] = f2bf(xh);
                } else {
                    float uo  = u[idx];
                    float fxu = val + uo;
                    float t   = fabsf(fxu) - THRESH;
                    t = t > 0.0f ? t : 0.0f;
                    float z  = copysignf(t, fxu);
                    float un = uo + fxu - z;
                    u[idx]     = un;
                    outbf[idx] = f2bf(z - un);   // (z - u) feeds next iteration's GEMM1
                }
            }
        }
    }
}

// ---------------------------------------------------------------------------
// Final clip to [min(x), max(x)]
// ---------------------------------------------------------------------------
__global__ void k_clip(const float* __restrict__ xh, const float* __restrict__ mnmx,
                       float* __restrict__ out, int n) {
    float mn = mnmx[0], mx = mnmx[1];
    for (int i = blockIdx.x * blockDim.x + threadIdx.x; i < n; i += gridDim.x * blockDim.x)
        out[i] = fminf(fmaxf(xh[i], mn), mx);
}

// ---------------------------------------------------------------------------
// Host-side orchestration
// ---------------------------------------------------------------------------
extern "C" void kernel_launch(void* const* d_in, const int* in_sizes, int n_in,
                              void* d_out, int out_size, void* d_ws, size_t ws_size,
                              hipStream_t stream) {
    const float* y   = (const float*)d_in[0];   // (B, M)
    const float* x   = (const float*)d_in[1];   // (B, A)
    const float* a   = (const float*)d_in[2];   // (M, A)
    const float* phi = (const float*)d_in[3];   // (S, A)
    float* out = (float*)d_out;                 // (B, A)

    char* ws = (char*)d_ws;
    size_t off = 0;
    auto carve = [&](size_t bytes) -> char* {
        char* p = ws + off;
        off = (off + bytes + 255) & ~(size_t)255;
        return p;
    };

    // u region is also reused as the Gram matrix m during setup (m: A*A*4 <= B*S*4)
    float*  u_buf   = (float*) carve((size_t)BATCH * SDIM * 4);   // u (and m in setup)
    float*  m_buf   = u_buf;
    float*  dU_buf  = (float*) carve((size_t)ADIM * 4);
    __bf16* aT_bf   = (__bf16*)carve((size_t)ADIM * MDIM * 2);    // a^T  (A, M)
    __bf16* y_bf    = (__bf16*)carve((size_t)BATCH * MDIM * 2);   // y    (B, M)
    __bf16* phi_bf  = (__bf16*)carve((size_t)SDIM * ADIM * 2);    // phi  (S, A)
    __bf16* phiT_bf = (__bf16*)carve((size_t)ADIM * SDIM * 2);    // phiT (A, S)
    float*  x0_buf  = (float*) carve((size_t)BATCH * ADIM * 4);   // x0   (B, A)
    float*  xh_f    = (float*) carve((size_t)BATCH * ADIM * 4);   // x_hat f32
    __bf16* xh_bf   = (__bf16*)carve((size_t)BATCH * ADIM * 2);   // x_hat bf16
    __bf16* zmu_bf  = (__bf16*)carve((size_t)BATCH * SDIM * 2);   // (z - u) bf16
    float*  pmin    = (float*) carve(256 * 4);
    float*  pmax    = (float*) carve(256 * 4);
    float*  mnmx    = (float*) carve(2 * 4);
    (void)ws_size; (void)in_sizes; (void)n_in; (void)out_size; (void)x;

    // ---- conversions / transposes ----
    {
        dim3 blk(32, 8);
        k_transpose_bf16<<<dim3(ADIM / 32, MDIM / 32), blk, 0, stream>>>(a,   aT_bf,   MDIM, ADIM);
        k_transpose_bf16<<<dim3(ADIM / 32, SDIM / 32), blk, 0, stream>>>(phi, phiT_bf, SDIM, ADIM);
    }
    k_cvt_bf16<<<2048, 256, 0, stream>>>(y,   y_bf,   BATCH * MDIM);
    k_cvt_bf16<<<2048, 256, 0, stream>>>(phi, phi_bf, SDIM * ADIM);

    // ---- min/max of x ----
    k_minmax_part <<<256, 256, 0, stream>>>(x, BATCH * ADIM, pmin, pmax);
    k_minmax_final<<<1,   256, 0, stream>>>(pmin, pmax, 256, mnmx);

    // ---- m = a^T a + RHO * phi^T phi  (Gram: A-op and Bt-op are the same buffer) ----
    k_gemm<0><<<dim3(ADIM / 128, ADIM / 128), 256, 0, stream>>>(
        aT_bf, aT_bf, m_buf, nullptr, nullptr, nullptr, nullptr, nullptr,
        ADIM, ADIM, MDIM, 1.0f, 0);
    k_gemm<0><<<dim3(ADIM / 128, ADIM / 128), 256, 0, stream>>>(
        phiT_bf, phiT_bf, m_buf, nullptr, nullptr, nullptr, nullptr, nullptr,
        ADIM, ADIM, SDIM, RHO_C, 1);

    // ---- dU = 1/diag(U) from unpivoted LU of m ----
    k_lu_diag<<<1, 1024, 0, stream>>>(m_buf, dU_buf, ADIM);

    // ---- x0 = y @ a   (Bt-op = a^T stored row-major) ----
    k_gemm<0><<<dim3(ADIM / 128, BATCH / 128), 256, 0, stream>>>(
        y_bf, aT_bf, x0_buf, nullptr, nullptr, nullptr, nullptr, nullptr,
        BATCH, ADIM, MDIM, 1.0f, 0);

    // ---- init ADMM state (after LU since u aliases m) ----
    k_zero_f32 <<<2048, 256, 0, stream>>>(u_buf,  BATCH * SDIM);
    k_zero_bf16<<<2048, 256, 0, stream>>>(zmu_bf, BATCH * SDIM);

    // ---- 10 ADMM iterations: two fused WMMA GEMMs each ----
    for (int it = 0; it < ITERS; ++it) {
        // x_hat = dU * (x0 + (z-u) @ (RHO*phi))   [out (B,A), K=S]
        k_gemm<1><<<dim3(ADIM / 128, BATCH / 128), 256, 0, stream>>>(
            zmu_bf, phiT_bf, nullptr, x0_buf, dU_buf, xh_f, xh_bf, nullptr,
            BATCH, ADIM, SDIM, RHO_C, 0);
        // fxu = x_hat @ phi^T + u; z = shrink(fxu); u += fxu - z; zmu = z - u
        k_gemm<2><<<dim3(SDIM / 128, BATCH / 128), 256, 0, stream>>>(
            xh_bf, phi_bf, nullptr, nullptr, nullptr, nullptr, zmu_bf, u_buf,
            BATCH, SDIM, ADIM, 1.0f, 0);
    }

    // ---- output: clip(x_hat, min(x), max(x)) ----
    k_clip<<<2048, 256, 0, stream>>>(xh_f, mnmx, out, BATCH * ADIM);
}
